// Model_82566451298376
// MI455X (gfx1250) — matrix operand order repaired and checked by reading, not compiled, not run on hardware
//
#include <hip/hip_runtime.h>
#include <hip/hip_bf16.h>

// ---------------------------------------------------------------------------
// Single-head attention (S = D = 4096), bf16 WMMA GEMMs, fp32 accumulate.
//   q = query@Wq^T+bq ; k = key@Wk^T+bk ; v = value@Wv^T+bv     (NT GEMMs)
//   scores = q@k^T / 64 ; attn = softmax(scores)
//   ctx[s,d] = sum_j attn[s,j]*v[d,j]  (== NT GEMM attn x v)
//   out = ctx@Wo^T + bo + value
// GEMM: 128x256 block, 8 waves, 64x64 wave tile (16 WMMA / K-step),
// double-buffered LDS, async global->LDS copies for bf16 operands.
// ---------------------------------------------------------------------------

#define SD  4096
#define BM  128
#define BN  256
#define BK  32
#define NK  (SD / BK)
#define LDT 40   // padded LDS row stride (elements): 80B

typedef __attribute__((ext_vector_type(16))) __bf16 v16bf;
typedef __attribute__((ext_vector_type(8)))  __bf16 bf16x8;
typedef __attribute__((ext_vector_type(8)))  float  v8f;

__device__ __forceinline__ __bf16 f2bf(float f) { return (__bf16)f; }

// Drain this wave's ASYNCcnt (async global->LDS copies) before barrier.
__device__ __forceinline__ void wait_async0() {
  asm volatile("s_wait_asynccnt 0x0" ::: "memory");
}

// Copy 8 K-contiguous elements into LDS.
//  bf16 source : one 16B async global->LDS copy (ASYNCcnt, no VGPR round trip)
//  fp32 source : 32B load + v_cvt_pk_bf16_f32 + ds_store_b128
template <typename T>
__device__ __forceinline__ void copy8(const T* __restrict__ g, __bf16* l) {
  if constexpr (sizeof(T) == 2) {
    unsigned lo = (unsigned)(size_t)l;   // low 32 bits of generic = LDS offset
    asm volatile("global_load_async_to_lds_b128 %0, %1, off"
                 :: "v"(lo), "v"(g) : "memory");
  } else {
    const float4* s = (const float4*)g;
    float4 x0 = s[0], x1 = s[1];
    bf16x8 a;
    a[0]=f2bf(x0.x); a[1]=f2bf(x0.y); a[2]=f2bf(x0.z); a[3]=f2bf(x0.w);
    a[4]=f2bf(x1.x); a[5]=f2bf(x1.y); a[6]=f2bf(x1.z); a[7]=f2bf(x1.w);
    *(bf16x8*)l = a;
  }
}

// Stage one K-tile: A[128x32] (2 chunks/thread), B[256x32] (4 chunks/thread).
template <typename TA, typename TB>
__device__ __forceinline__ void stage_tiles(const TA* __restrict__ A,
                                            const TB* __restrict__ B,
                                            __bf16* la, __bf16* lb,
                                            int bm, int bn, int k0, int t) {
#pragma unroll
  for (int i = 0; i < 2; ++i) {
    const int c = t + i * 256;          // 512 chunks of 8
    const int row = c >> 2, co = (c & 3) * 8;
    copy8<TA>(A + (size_t)(bm + row) * SD + k0 + co, la + row * LDT + co);
  }
#pragma unroll
  for (int i = 0; i < 4; ++i) {
    const int c = t + i * 256;          // 1024 chunks of 8
    const int row = c >> 2, co = (c & 3) * 8;
    copy8<TB>(B + (size_t)(bn + row) * SD + k0 + co, lb + row * LDT + co);
  }
}

// C[M,N] = scale * (A[M,K] x B[N,K]^T) + bias[n] + residual[m,n]
template <typename TA, typename TB, bool OUT_BF16>
__global__ __launch_bounds__(256) void gemm_nt(const TA* __restrict__ A,
                                               const TB* __restrict__ B,
                                               const float* __restrict__ bias,
                                               const float* __restrict__ residual,
                                               void* __restrict__ Cout,
                                               float scale) {
  __shared__ __bf16 lds_a[2][BM * LDT];   // 2 x 10 KB
  __shared__ __bf16 lds_b[2][BN * LDT];   // 2 x 20 KB

  const int t     = threadIdx.x;
  const int lane  = t & 31;
  const int wid   = t >> 5;       // 0..7
  const int wm    = wid >> 2;     // 0..1 : 64-row band
  const int wn    = wid & 3;      // 0..3 : 64-col band
  const int lrow  = lane & 15;
  const int lhalf = lane >> 4;

  const int bm = blockIdx.y * BM;
  const int bn = blockIdx.x * BN;

  v8f acc[4][4] = {};

  // Prologue: stage tile 0 into buffer 0.
  stage_tiles<TA, TB>(A, B, lds_a[0], lds_b[0], bm, bn, 0, t);
  wait_async0();
  __syncthreads();

  for (int kt = 0; kt < NK; ++kt) {
    const int buf = kt & 1;

    // Stage next tile into the other buffer; overlaps with WMMA below.
    if (kt + 1 < NK)
      stage_tiles<TA, TB>(A, B, lds_a[buf ^ 1], lds_b[buf ^ 1],
                          bm, bn, (kt + 1) * BK, t);

    // A fragment (16x32): lane half h holds K in [16h, 16h+16)
    v16bf afr[4];
#pragma unroll
    for (int ti = 0; ti < 4; ++ti) {
      const int m = wm * 64 + ti * 16 + lrow;
      union { v16bf v; bf16x8 h[2]; } u;
      u.h[0] = *(const bf16x8*)&lds_a[buf][m * LDT + lhalf * 16];
      u.h[1] = *(const bf16x8*)&lds_a[buf][m * LDT + lhalf * 16 + 8];
      afr[ti] = u.v;
    }
    // B fragment (32x16): lane half h holds K in [16h, 16h+16)
    v16bf bfr[4];
#pragma unroll
    for (int tj = 0; tj < 4; ++tj) {
      const int n = wn * 64 + tj * 16 + lrow;
      union { v16bf v; bf16x8 h[2]; } u;
      u.h[0] = *(const bf16x8*)&lds_b[buf][n * LDT + lhalf * 16];
      u.h[1] = *(const bf16x8*)&lds_b[buf][n * LDT + lhalf * 16 + 8];
      bfr[tj] = u.v;
    }

#pragma unroll
    for (int ti = 0; ti < 4; ++ti)
#pragma unroll
      for (int tj = 0; tj < 4; ++tj)
        acc[ti][tj] = __builtin_amdgcn_wmma_f32_16x16x32_bf16(
            false, afr[ti], false, bfr[tj], (short)0, acc[ti][tj], false, false);

    wait_async0();     // this wave's async copies for tile kt+1 are in LDS
    __syncthreads();   // all waves done computing buf & staging buf^1
  }

  // Epilogue.  D layout: vgpr v -> M = 8*lhalf + v ; N = lrow.
#pragma unroll
  for (int ti = 0; ti < 4; ++ti) {
#pragma unroll
    for (int tj = 0; tj < 4; ++tj) {
      const int col  = bn + wn * 64 + tj * 16 + lrow;
      const float bc = bias ? bias[col] : 0.0f;
#pragma unroll
      for (int v = 0; v < 8; ++v) {
        const int row = bm + wm * 64 + ti * 16 + lhalf * 8 + v;
        float val = acc[ti][tj][v] * scale + bc;
        if (residual) val += residual[(size_t)row * SD + col];
        if constexpr (OUT_BF16)
          ((__bf16*)Cout)[(size_t)row * SD + col] = f2bf(val);
        else
          ((float*)Cout)[(size_t)row * SD + col] = val;
      }
    }
  }
}

// Row softmax: fp32 scores in, bf16 probabilities out. One 256-thread WG/row.
__global__ __launch_bounds__(256) void softmax_row(const float* __restrict__ S,
                                                   __bf16* __restrict__ P) {
  __shared__ float red[256];
  const int row = blockIdx.x;
  const int t   = threadIdx.x;
  const float* p = S + (size_t)row * SD;

  float x[16];
  float mx = -3.0e38f;
#pragma unroll
  for (int i = 0; i < 16; ++i) { x[i] = p[t + 256 * i]; mx = fmaxf(mx, x[i]); }
  red[t] = mx; __syncthreads();
  for (int s = 128; s > 0; s >>= 1) {
    if (t < s) red[t] = fmaxf(red[t], red[t + s]);
    __syncthreads();
  }
  mx = red[0]; __syncthreads();

  float sum = 0.0f;
#pragma unroll
  for (int i = 0; i < 16; ++i) { x[i] = __expf(x[i] - mx); sum += x[i]; }
  red[t] = sum; __syncthreads();
  for (int s = 128; s > 0; s >>= 1) {
    if (t < s) red[t] += red[t + s];
    __syncthreads();
  }
  const float inv = 1.0f / red[0];

  __bf16* q = P + (size_t)row * SD;
#pragma unroll
  for (int i = 0; i < 16; ++i) q[t + 256 * i] = f2bf(x[i] * inv);
}

extern "C" void kernel_launch(void* const* d_in, const int* in_sizes, int n_in,
                              void* d_out, int out_size, void* d_ws, size_t ws_size,
                              hipStream_t stream) {
  (void)in_sizes; (void)n_in; (void)out_size; (void)ws_size;

  const float* query = (const float*)d_in[0];
  const float* key   = (const float*)d_in[1];
  const float* value = (const float*)d_in[2];
  const float* Wq    = (const float*)d_in[3];
  const float* bq    = (const float*)d_in[4];
  const float* Wk    = (const float*)d_in[5];
  const float* bk    = (const float*)d_in[6];
  const float* Wv    = (const float*)d_in[7];
  const float* bv    = (const float*)d_in[8];
  const float* Wo    = (const float*)d_in[9];
  const float* bo    = (const float*)d_in[10];
  float* out = (float*)d_out;

  const size_t MAT_BF = (size_t)SD * SD * sizeof(__bf16);  // 32 MiB
  char* w = (char*)d_ws;
  __bf16* qb   = (__bf16*)(w + 0 * MAT_BF);
  __bf16* kb   = (__bf16*)(w + 1 * MAT_BF);
  __bf16* vb   = (__bf16*)(w + 2 * MAT_BF);
  __bf16* attn = (__bf16*)(w + 3 * MAT_BF);
  __bf16* ctxb = (__bf16*)(w + 4 * MAT_BF);
  float*  sc   = (float*) (w + 5 * MAT_BF);                // 64 MiB fp32 scores

  dim3 grid(SD / BN, SD / BM);
  dim3 blk(256);

  // Projections: fp32 in -> bf16 out (bias fused)
  gemm_nt<float, float, true ><<<grid, blk, 0, stream>>>(query, Wq, bq, nullptr, qb, 1.0f);
  gemm_nt<float, float, true ><<<grid, blk, 0, stream>>>(key,   Wk, bk, nullptr, kb, 1.0f);
  gemm_nt<float, float, true ><<<grid, blk, 0, stream>>>(value, Wv, bv, nullptr, vb, 1.0f);
  // scores = q k^T / 64  (fp32 out)
  gemm_nt<__bf16, __bf16, false><<<grid, blk, 0, stream>>>(qb, kb, nullptr, nullptr, sc, 1.0f / 64.0f);
  // attn = softmax(scores)  (bf16 out)
  softmax_row<<<dim3(SD), blk, 0, stream>>>(sc, attn);
  // ctx = attn v^T  (NT GEMM against row-major v)
  gemm_nt<__bf16, __bf16, true ><<<grid, blk, 0, stream>>>(attn, vb, nullptr, nullptr, ctxb, 1.0f);
  // out = ctx Wo^T + bo + value  (fp32 out, residual fused)
  gemm_nt<__bf16, float, false><<<grid, blk, 0, stream>>>(ctxb, Wo, bo, value, out, 1.0f);
}
